// RFND_45938970198744
// MI455X (gfx1250) — matrix operand order repaired and checked by reading, not compiled, not run on hardware
//
#include <hip/hip_runtime.h>

// ---------------- constants (from reference) ----------------
enum { DD = 200, KK = 10, NHEAD = 5, HDIM = 40, ANSN = 2, GHEADN = 5, GCN = 2,
       BB = 8, LL = 80, RR = 64, H1 = 400, PF = 800, HPAD = 416, NP200 = 208 };

// ---------------- WMMA helpers (gfx1250, wave32) ----------------
typedef __bf16  v16bf __attribute__((ext_vector_type(16)));
typedef __bf16  v8bf  __attribute__((ext_vector_type(8)));
typedef float   v8f   __attribute__((ext_vector_type(8)));
typedef float   v4f   __attribute__((ext_vector_type(4)));

static __device__ inline __bf16 tobf(float x) { return (__bf16)x; }

static __device__ inline v8f wmma_bf16(v16bf a, v16bf b, v8f c) {
  return __builtin_amdgcn_wmma_f32_16x16x32_bf16(false, a, false, b, (short)0, c, false, false);
}

static __device__ inline v16bf cat16(v8bf lo, v8bf hi) {
  return __builtin_shufflevector(lo, hi, 0,1,2,3,4,5,6,7,8,9,10,11,12,13,14,15);
}

// ---------------- weight packer: W (K x N f32, ld=N) -> fragment-ordered bf16
// out[((kb*Np + n)*32) + t] = bf16(W[(kb*32+kt(t))*N + n]), kt(t) = (t&7)|((t&8)<<1)|((t&16)>>1), zero beyond K/N.
// grid: (Np, ceil(K/32)), block: 32
__global__ void k_packW(const float* __restrict__ W, int K, int N, int Np,
                        __bf16* __restrict__ out)
{
  int n = blockIdx.x, kb = blockIdx.y, t = threadIdx.x;
  int kk = kb * 32 + ((t & 7) | ((t & 8) << 1) | ((t & 16) >> 1));
  float v = (kk < K && n < N) ? W[(long)kk * N + n] : 0.f;
  out[((long)kb * Np + n) * 32 + t] = tobf(v);
}

// ---------------- generic WMMA GEMM: C = act(A @ W + bias) ----------------
// A: (grid.z batches) M x K f32, row stride lda, batch stride sA.
// Wp: packed bf16 (Np cols, K padded to mult of 32). C: M x N_, row stride ldc.
__global__ __launch_bounds__(32)
void k_gemm(const float* __restrict__ A, long sA, int lda,
            const __bf16* __restrict__ Wp, int Np,
            const float* __restrict__ bias,
            float* __restrict__ C, long sC, int ldc,
            int M, int K, int N_, int act)
{
  int lane = threadIdx.x;
  int m0 = blockIdx.x * 16, n0 = blockIdx.y * 16;
  const float* Ab = A + (long)blockIdx.z * sA;
  float* Cb = C + (long)blockIdx.z * sC;
  int kh = lane >> 4, nl = lane & 15;
  int mrow = m0 + nl;
  int ar = (mrow < M) ? mrow : (M - 1);   // clamp: OOB rows are never stored
  int ncol = n0 + nl;                     // < Np by construction
  v8f acc = {0.f,0.f,0.f,0.f,0.f,0.f,0.f,0.f};
  int KB = K >> 5;
  const float* arow = Ab + (long)ar * lda;
  for (int kb = 0; kb < KB; ++kb) {
    int k0 = kb * 32;
    const float* ap = arow + k0 + kh * 8;
    __builtin_prefetch(ap + 32, 0, 1);
    v4f a0 = *(const v4f*)ap;
    v4f a1 = *(const v4f*)(ap + 4);
    v4f a2 = *(const v4f*)(ap + 16);
    v4f a3 = *(const v4f*)(ap + 20);
    v16bf af;
#pragma unroll
    for (int j = 0; j < 4; ++j) {
      af[j]      = tobf(a0[j]);
      af[j + 4]  = tobf(a1[j]);
      af[j + 8]  = tobf(a2[j]);
      af[j + 12] = tobf(a3[j]);
    }
    v16bf bf_ = *(const v16bf*)(Wp + ((long)kb * Np + ncol) * 32 + kh * 16);
    acc = wmma_bf16(af, bf_, acc);
  }
  if (K & 31) {                            // guarded remainder (selects, no branches)
    int k0 = KB * 32;
    v16bf af;
#pragma unroll
    for (int j = 0; j < 8; ++j) {
      int kk0 = k0 + kh * 8 + j;
      int kk1 = kk0 + 16;
      int kc0 = (kk0 < K) ? kk0 : (K - 1);
      int kc1 = (kk1 < K) ? kk1 : (K - 1);
      float v0 = arow[kc0];
      float v1 = arow[kc1];
      af[j]     = tobf((kk0 < K) ? v0 : 0.f);
      af[j + 8] = tobf((kk1 < K) ? v1 : 0.f);
    }
    v16bf bf_ = *(const v16bf*)(Wp + ((long)KB * Np + ncol) * 32 + kh * 16);
    acc = wmma_bf16(af, bf_, acc);         // packed weights are zero beyond K
  }
#pragma unroll
  for (int i = 0; i < 8; ++i) {
    int row = m0 + i + 8 * kh;
    if (row < M && ncol < N_) {
      float v = acc[i] + (bias ? bias[ncol] : 0.f);
      if (act == 1) v = fmaxf(v, 0.f);
      Cb[(long)row * ldc + ncol] = v;
    }
  }
}

// ---------------- fused pair-MLP kernels ----------------
// feature = [a, b, a*b, a-b] (len 800) built in LDS as bf16; layer1 800->400
// (relu), layer2 400->200 (relu). Weights pre-packed bf16.
__global__ __launch_bounds__(32)
void k_pair_score(const float* abase, long sa, const float* bbase, long sb,
                  int n1, int m1,
                  const __bf16* __restrict__ pW1, const float* __restrict__ b1,
                  const __bf16* __restrict__ pW2, const float* __restrict__ b2,
                  const float* __restrict__ wtop, const float* btop_all, int ti,
                  float* __restrict__ score)
{
  __shared__ __align__(16) __bf16 sfeat[16][PF];
  __shared__ __align__(16) __bf16 shid[16][HPAD];
  __shared__ float sacc[16];
  int lane = threadIdx.x;
  int b = blockIdx.z;
  int np = n1 * m1;
  int p0 = blockIdx.x * 16;
  for (int idx = lane; idx < 16 * DD; idx += 32) {
    int r = idx / DD, c = idx % DD;
    int p = p0 + r;
    int pc = (p < np) ? p : 0;
    int i = pc / m1, j = pc % m1;
    float av = abase[(long)b * sa + (long)i * DD + c];
    float bv = bbase[(long)b * sb + (long)j * DD + c];
    sfeat[r][c]          = tobf(av);
    sfeat[r][DD + c]     = tobf(bv);
    sfeat[r][2 * DD + c] = tobf(av * bv);
    sfeat[r][3 * DD + c] = tobf(av - bv);
  }
  if (lane < 16) {
    sacc[lane] = 0.f;
    for (int r = 0; r < 16; ++r) shid[r][H1 + lane] = tobf(0.f);
  }
  __syncthreads();
  int kh = lane >> 4, nl = lane & 15;
  // layer 1: 16x400, K=800
  for (int nt = 0; nt < 25; ++nt) {
    int n0 = nt * 16;
    v8f acc = {0.f,0.f,0.f,0.f,0.f,0.f,0.f,0.f};
    for (int kb = 0; kb < 25; ++kb) {
      int k0 = kb * 32;
      v8bf lo = *(const v8bf*)&sfeat[nl][k0 + kh * 8];
      v8bf hi = *(const v8bf*)&sfeat[nl][k0 + 16 + kh * 8];
      v16bf bf_ = *(const v16bf*)(pW1 + ((long)kb * H1 + (n0 + nl)) * 32 + kh * 16);
      acc = wmma_bf16(cat16(lo, hi), bf_, acc);
    }
    int col = n0 + nl;
#pragma unroll
    for (int i = 0; i < 8; ++i)
      shid[i + 8 * kh][col] = tobf(fmaxf(acc[i] + b1[col], 0.f));
  }
  __syncthreads();
  // layer 2: 16x200, K=416 (hidden + weights both zero-padded)
  for (int nt = 0; nt < 13; ++nt) {
    int n0 = nt * 16;
    v8f acc = {0.f,0.f,0.f,0.f,0.f,0.f,0.f,0.f};
    for (int kb = 0; kb < 13; ++kb) {
      int k0 = kb * 32;
      v8bf lo = *(const v8bf*)&shid[nl][k0 + kh * 8];
      v8bf hi = *(const v8bf*)&shid[nl][k0 + 16 + kh * 8];
      v16bf bf_ = *(const v16bf*)(pW2 + ((long)kb * NP200 + (n0 + nl)) * 32 + kh * 16);
      acc = wmma_bf16(cat16(lo, hi), bf_, acc);
    }
    int col = n0 + nl;
    if (col < DD) {
#pragma unroll
      for (int i = 0; i < 8; ++i) {
        float e = fmaxf(acc[i] + b2[col], 0.f);
        atomicAdd(&sacc[i + 8 * kh], e * wtop[col]);
      }
    }
  }
  __syncthreads();
  if (lane < 16) {
    int p = p0 + lane;
    if (p < np) score[(long)b * np + p] = fmaxf(sacc[lane] + btop_all[ti], 0.f);
  }
}

__global__ __launch_bounds__(32)
void k_pair_rows(const float* abase, long sa, const float* bbase, long sb,
                 int m1, const int* __restrict__ selidx,
                 const __bf16* __restrict__ pW1, const float* __restrict__ b1,
                 const __bf16* __restrict__ pW2, const float* __restrict__ b2,
                 float* __restrict__ Ecat, int off)
{
  __shared__ __align__(16) __bf16 sfeat[16][PF];
  __shared__ __align__(16) __bf16 shid[16][HPAD];
  int lane = threadIdx.x;
  int b = blockIdx.x;
  for (int idx = lane; idx < 16 * DD; idx += 32) {
    int r = idx / DD, c = idx % DD;
    int p = (r < KK) ? selidx[b * KK + r] : 0;
    int i = p / m1, j = p % m1;
    float av = abase[(long)b * sa + (long)i * DD + c];
    float bv = bbase[(long)b * sb + (long)j * DD + c];
    sfeat[r][c]          = tobf(av);
    sfeat[r][DD + c]     = tobf(bv);
    sfeat[r][2 * DD + c] = tobf(av * bv);
    sfeat[r][3 * DD + c] = tobf(av - bv);
  }
  if (lane < 16)
    for (int r = 0; r < 16; ++r) shid[r][H1 + lane] = tobf(0.f);
  __syncthreads();
  int kh = lane >> 4, nl = lane & 15;
  for (int nt = 0; nt < 25; ++nt) {
    int n0 = nt * 16;
    v8f acc = {0.f,0.f,0.f,0.f,0.f,0.f,0.f,0.f};
    for (int kb = 0; kb < 25; ++kb) {
      int k0 = kb * 32;
      v8bf lo = *(const v8bf*)&sfeat[nl][k0 + kh * 8];
      v8bf hi = *(const v8bf*)&sfeat[nl][k0 + 16 + kh * 8];
      v16bf bf_ = *(const v16bf*)(pW1 + ((long)kb * H1 + (n0 + nl)) * 32 + kh * 16);
      acc = wmma_bf16(cat16(lo, hi), bf_, acc);
    }
    int col = n0 + nl;
#pragma unroll
    for (int i = 0; i < 8; ++i)
      shid[i + 8 * kh][col] = tobf(fmaxf(acc[i] + b1[col], 0.f));
  }
  __syncthreads();
  for (int nt = 0; nt < 13; ++nt) {
    int n0 = nt * 16;
    v8f acc = {0.f,0.f,0.f,0.f,0.f,0.f,0.f,0.f};
    for (int kb = 0; kb < 13; ++kb) {
      int k0 = kb * 32;
      v8bf lo = *(const v8bf*)&shid[nl][k0 + kh * 8];
      v8bf hi = *(const v8bf*)&shid[nl][k0 + 16 + kh * 8];
      v16bf bf_ = *(const v16bf*)(pW2 + ((long)kb * NP200 + (n0 + nl)) * 32 + kh * 16);
      acc = wmma_bf16(cat16(lo, hi), bf_, acc);
    }
    int col = n0 + nl;
#pragma unroll
    for (int i = 0; i < 8; ++i) {
      int row = i + 8 * kh;
      if (col < DD && row < KK) {
        float e = fmaxf(acc[i] + b2[col], 0.f);
        Ecat[((long)b * 50 + off + row) * DD + col] = e;
      }
    }
  }
}

// ---------------- small scalar kernels ----------------
__global__ void k_rowdot_relu(const float* __restrict__ E, const float* __restrict__ w,
                              const float* ball, int bi, float* __restrict__ out)
{
  __shared__ float red[64];
  int r = blockIdx.x, t = threadIdx.x;
  const float* row = E + (long)r * DD;
  float s = 0.f;
  for (int c = t; c < DD; c += 64) s += row[c] * w[c];
  red[t] = s; __syncthreads();
  for (int o = 32; o > 0; o >>= 1) { if (t < o) red[t] += red[t + o]; __syncthreads(); }
  if (t == 0) out[r] = fmaxf(red[0] + ball[bi], 0.f);
}

__global__ void k_topk(const float* __restrict__ score, const unsigned char* mask,
                       int len, int* __restrict__ idx_out)
{
  __shared__ float bv[256]; __shared__ int bi_[256]; __shared__ int chosen[KK];
  int b = blockIdx.x, t = threadIdx.x;
  const float* s = score + (long)b * len;
  const unsigned char* mk = mask ? mask + (long)b * len : nullptr;
  for (int k = 0; k < KK; ++k) {
    float best = -3.0e38f; int bidx = -1;
    for (int p = t; p < len; p += 256) {
      if (mk && mk[p] == 0) continue;
      bool used = false;
      for (int q = 0; q < k; ++q) if (chosen[q] == p) used = true;
      if (used) continue;
      float v = s[p];
      if (v > best) { best = v; bidx = p; }
    }
    bv[t] = best; bi_[t] = bidx; __syncthreads();
    for (int o = 128; o > 0; o >>= 1) {
      if (t < o && bv[t + o] > bv[t]) { bv[t] = bv[t + o]; bi_[t] = bi_[t + o]; }
      __syncthreads();
    }
    if (t == 0) { int ci = (bi_[0] < 0) ? 0 : bi_[0]; chosen[k] = ci; idx_out[b * KK + k] = ci; }
    __syncthreads();
  }
}

__global__ void k_gather(const float* __restrict__ E, int rowsPerB,
                         const int* __restrict__ idx, float* __restrict__ Ecat, int off)
{
  int b = blockIdx.x / KK, k = blockIdx.x % KK;
  int sel = idx[b * KK + k];
  const float* src = E + ((long)b * rowsPerB + sel) * DD;
  float* dst = Ecat + ((long)b * 50 + off + k) * DD;
  for (int c = threadIdx.x; c < DD; c += 64) dst[c] = src[c];
}

__global__ void k_bind(const float* __restrict__ Ecat, const float* __restrict__ clues,
                       float* __restrict__ bound)
{
  __shared__ float red[64]; __shared__ float att[5];
  int row = blockIdx.x, t = threadIdx.x;
  int n = row % 50, ty = n / KK;
  const float* e = Ecat + (long)row * DD;
  const float* cl = clues + (long)ty * 5 * DD;
  for (int c5 = 0; c5 < 5; ++c5) {
    float s = 0.f;
    for (int c = t; c < DD; c += 64) s += e[c] * cl[c5 * DD + c];
    red[t] = s; __syncthreads();
    for (int o = 32; o > 0; o >>= 1) { if (t < o) red[t] += red[t + o]; __syncthreads(); }
    if (t == 0) att[c5] = red[0];
    __syncthreads();
  }
  if (t == 0) {
    float mx = att[0];
    for (int c5 = 1; c5 < 5; ++c5) mx = fmaxf(mx, att[c5]);
    float ss = 0.f;
    for (int c5 = 0; c5 < 5; ++c5) { att[c5] = __expf(att[c5] - mx); ss += att[c5]; }
    for (int c5 = 0; c5 < 5; ++c5) att[c5] /= ss;
  }
  __syncthreads();
  for (int c = t; c < DD; c += 64) {
    float v = 0.f;
    for (int c5 = 0; c5 < 5; ++c5) v += att[c5] * cl[c5 * DD + c];
    bound[(long)row * DD + c] = v;
  }
}

__global__ void k_preds(const float* __restrict__ lb, const float* __restrict__ ans1,
                        float* __restrict__ preds)
{
  int row = blockIdx.x, t = threadIdx.x;
  for (int idx = t; idx < 2 * DD; idx += 256) {
    int a = idx / DD, c = idx % DD;
    preds[(long)row * 2 * DD + a * DD + c] = fmaxf(lb[(long)row * DD + c] + ans1[a * DD + c], 0.f);
  }
}

__global__ void k_inst(const float* __restrict__ il, const float* __restrict__ ans2,
                       float* __restrict__ isc)
{
  __shared__ float red[64];
  int row = blockIdx.x, t = threadIdx.x;
  for (int a = 0; a < ANSN; ++a) {
    float s = 0.f;
    for (int c = t; c < DD; c += 64) s += il[(long)row * DD + c] * ans2[a * DD + c];
    red[t] = s; __syncthreads();
    for (int o = 32; o > 0; o >>= 1) { if (t < o) red[t] += red[t + o]; __syncthreads(); }
    if (t == 0) isc[row * 2 + a] = red[0];
    __syncthreads();
  }
}

__global__ void k_attn(const float* __restrict__ q, const float* __restrict__ k,
                       const float* __restrict__ v, float* __restrict__ o)
{
  int b = blockIdx.x / NHEAD, h = blockIdx.x % NHEAD;
  int n = threadIdx.x;
  if (n >= 50) return;
  const float scale = 0.158113883f;  // 1/sqrt(40)
  float s[50];
  const float* qr = q + ((long)b * 50 + n) * DD + h * HDIM;
  float mx = -3e38f;
  for (int m = 0; m < 50; ++m) {
    const float* kr = k + ((long)b * 50 + m) * DD + h * HDIM;
    float d = 0.f;
    for (int c = 0; c < HDIM; ++c) d += qr[c] * kr[c];
    d *= scale; s[m] = d; mx = fmaxf(mx, d);
  }
  float ss = 0.f;
  for (int m = 0; m < 50; ++m) { s[m] = __expf(s[m] - mx); ss += s[m]; }
  float inv = 1.f / ss;
  for (int c = 0; c < HDIM; ++c) {
    float acc = 0.f;
    for (int m = 0; m < 50; ++m) acc += s[m] * v[((long)b * 50 + m) * DD + h * HDIM + c];
    o[((long)b * 50 + n) * DD + h * HDIM + c] = acc * inv;
  }
}

__global__ void k_ln(const float* __restrict__ X, int ldx, const float* __restrict__ Rr,
                     int ldr, float* __restrict__ out, int ldo,
                     const float* __restrict__ g, const float* __restrict__ bb)
{
  __shared__ float red[64]; __shared__ float mean_, var_;
  int r = blockIdx.x, t = threadIdx.x;
  const float* x = X + (long)r * ldx;
  const float* rr = Rr + (long)r * ldr;
  float s = 0.f;
  for (int c = t; c < DD; c += 64) s += x[c] + rr[c];
  red[t] = s; __syncthreads();
  for (int o = 32; o > 0; o >>= 1) { if (t < o) red[t] += red[t + o]; __syncthreads(); }
  if (t == 0) mean_ = red[0] / DD;
  __syncthreads();
  float m = mean_;
  s = 0.f;
  for (int c = t; c < DD; c += 64) { float d = x[c] + rr[c] - m; s += d * d; }
  red[t] = s; __syncthreads();
  for (int o = 32; o > 0; o >>= 1) { if (t < o) red[t] += red[t + o]; __syncthreads(); }
  if (t == 0) var_ = red[0] / DD;
  __syncthreads();
  float inv = rsqrtf(var_ + 1e-5f);
  for (int c = t; c < DD; c += 64)
    out[(long)r * ldo + c] = (x[c] + rr[c] - m) * inv * g[c] + bb[c];
}

__global__ void k_atoms(const float* __restrict__ Ecat, const float* __restrict__ preds,
                        const float* __restrict__ W1, const float* __restrict__ b1,
                        const float* __restrict__ W2, const float* __restrict__ b2s,
                        float* __restrict__ atoms)
{
  __shared__ float red[256];
  int rid = blockIdx.x;
  int bn = rid >> 1, a = rid & 1;
  int t = threadIdx.x;
  const float* e = Ecat + (long)bn * DD;
  const float* p = preds + (long)bn * 2 * DD + a * DD;
  float sacc = 0.f;
  for (int j = t; j < H1; j += 256) {
    float h = b1[j];
    for (int c = 0; c < DD; ++c) {
      float ec = e[c], pc = p[c];
      h += ec * W1[(long)c * H1 + j];
      h += pc * W1[(long)(c + DD) * H1 + j];
      h += (ec - pc) * W1[(long)(c + 2 * DD) * H1 + j];
      h += (ec * pc) * W1[(long)(c + 3 * DD) * H1 + j];
    }
    sacc += fmaxf(h, 0.f) * W2[j];
  }
  red[t] = sacc; __syncthreads();
  for (int o = 128; o > 0; o >>= 1) { if (t < o) red[t] += red[t + o]; __syncthreads(); }
  if (t == 0) {
    float s = red[0] + b2s[0];
    atoms[(long)a * 400 + bn] = 1.f / (1.f + __expf(-s));
  }
}

__global__ void k_softemb(const float* __restrict__ graph, int rowoff, int nrows,
                          const float* __restrict__ w, const float* bsc,
                          float* __restrict__ emb)
{
  __shared__ float lg[LL]; __shared__ float norm_;
  int b = blockIdx.x, t = threadIdx.x;
  const float* base = graph + ((long)b * (LL + RR) + rowoff) * DD;
  for (int r = t; r < nrows; r += 128) {
    float s = bsc[0];
    for (int c = 0; c < DD; ++c) s += base[(long)r * DD + c] * w[c];
    lg[r] = s;
  }
  __syncthreads();
  if (t == 0) {
    float mx = -3e38f;
    for (int r = 0; r < nrows; ++r) mx = fmaxf(mx, lg[r]);
    float ss = 0.f;
    for (int r = 0; r < nrows; ++r) { lg[r] = __expf(lg[r] - mx); ss += lg[r]; }
    norm_ = 1.f / ss;
  }
  __syncthreads();
  float nv = norm_;
  for (int c = t; c < DD; c += 128) {
    float acc = 0.f;
    for (int r = 0; r < nrows; ++r) acc += lg[r] * base[(long)r * DD + c];
    emb[(long)b * DD + c] = acc * nv;
  }
}

__global__ void k_eit(const float* __restrict__ Temb, const float* __restrict__ Iemb,
                      const float* __restrict__ W1, const float* __restrict__ b1,
                      const float* __restrict__ W2, const float* __restrict__ b2,
                      float* __restrict__ EIT)
{
  __shared__ float hid[H1]; __shared__ float tf[DD]; __shared__ float iff[DD];
  int b = blockIdx.x, t = threadIdx.x;
  for (int c = t; c < DD; c += 256) { tf[c] = Temb[(long)b * DD + c]; iff[c] = Iemb[(long)b * DD + c]; }
  __syncthreads();
  for (int j = t; j < H1; j += 256) {
    float h = b1[j];
    for (int c = 0; c < DD; ++c) {
      float a = tf[c], bv = iff[c];
      h += a * W1[(long)c * H1 + j];
      h += bv * W1[(long)(c + DD) * H1 + j];
      h += (a * bv) * W1[(long)(c + 2 * DD) * H1 + j];
      h += (a - bv) * W1[(long)(c + 3 * DD) * H1 + j];
    }
    hid[j] = fmaxf(h, 0.f);
  }
  __syncthreads();
  for (int c = t; c < DD; c += 256) {
    float s = b2[c];
    for (int j = 0; j < H1; ++j) s += hid[j] * W2[(long)j * DD + c];
    EIT[(long)b * DD + c] = fmaxf(s, 0.f);
  }
}

static __device__ inline void sparsemax2(float z0, float z1, float* o0, float* o1) {
  float hi = fmaxf(z0, z1), lo = fminf(z0, z1);
  float tau = (hi - lo < 1.f) ? (hi + lo - 1.f) * 0.5f : (hi - 1.f);
  *o0 = fmaxf(z0 - tau, 0.f);
  *o1 = fmaxf(z1 - tau, 0.f);
}

__global__ void k_heads(const float* __restrict__ preds2, const float* __restrict__ Wg,
                        const float* __restrict__ bg, const float* __restrict__ EIT,
                        const float* __restrict__ isc, const float* __restrict__ atoms,
                        float* __restrict__ outp)
{
  __shared__ float s_isp[100], s_g1[100], s_fcs0[100];
  __shared__ float s_g2, s_hv[2];
  int b = blockIdx.x, t = threadIdx.x;
  if (t < 50) {
    float z0 = isc[((long)b * 50 + t) * 2 + 0];
    float z1 = isc[((long)b * 50 + t) * 2 + 1];
    sparsemax2(z0, z1, &s_isp[t * 2], &s_isp[t * 2 + 1]);
  }
  if (t < 2) s_hv[t] = -3e38f;
  __syncthreads();
  for (int gi = 0; gi < GHEADN; ++gi) {
    const float* w1 = Wg + (long)(2 * gi) * DD;
    const float* w2 = Wg + (long)(2 * gi + 1) * DD;
    if (t == 0) {
      float s = bg[2 * gi + 1];
      for (int c = 0; c < DD; ++c) s += EIT[(long)b * DD + c] * w2[c];
      s_g2 = s;
    }
    for (int idx = t; idx < 100; idx += 64) {
      int n = idx >> 1, a = idx & 1;
      const float* pr = preds2 + ((long)(b * 50 + n) * 2 + a) * DD;
      float s = bg[2 * gi];
      for (int c = 0; c < DD; ++c) s += pr[c] * w1[c];
      s_g1[idx] = s;
    }
    __syncthreads();
    if (t < 50) {
      float o0, o1;
      sparsemax2(s_g1[t * 2] * s_g2, s_g1[t * 2 + 1] * s_g2, &o0, &o1);
      s_fcs0[t * 2]     = o0 * s_isp[t * 2];
      s_fcs0[t * 2 + 1] = o1 * s_isp[t * 2 + 1];
    }
    __syncthreads();
    if (t < 2) {
      int a = t;
      float z[50]; int id[50];
      for (int n = 0; n < 50; ++n) { z[n] = s_fcs0[n * 2 + a]; id[n] = n; }
      for (int i2 = 1; i2 < 50; ++i2) {
        float key = z[i2]; int kid = id[i2]; int j2 = i2 - 1;
        while (j2 >= 0 && z[j2] < key) { z[j2 + 1] = z[j2]; id[j2 + 1] = id[j2]; --j2; }
        z[j2 + 1] = key; id[j2 + 1] = kid;
      }
      float cs = 0.f, cssel = 0.f; int supp = 1;
      for (int k2 = 1; k2 <= 50; ++k2) {
        cs += z[k2 - 1];
        if (1.f + (float)k2 * z[k2 - 1] > cs) { supp = k2; cssel = cs; }
      }
      float tau = (cssel - 1.f) / (float)supp;
      const float* at = atoms + (long)a * 400 + b * 50;
      float mprod = 1.f, fb = 1.f; bool anyflag = false;
      for (int k2 = 0; k2 < 50; ++k2) {
        float f = fmaxf(z[k2] - tau, 0.f);
        if (f >= 0.4f) { anyflag = true; mprod *= at[id[k2]]; }
        if (k2 < 3) fb *= at[id[k2]] * f;
      }
      float head = anyflag ? mprod : fb;
      s_hv[a] = fmaxf(s_hv[a], head);
    }
    __syncthreads();
  }
  if (t < 2) outp[t * 8 + b] = s_hv[t];
}

__global__ void k_adjmm(const float* __restrict__ adj, const float* __restrict__ graph,
                        float* __restrict__ AG)
{
  int row = blockIdx.x;                 // b*144 + n
  int b = row / (LL + RR);
  int t = threadIdx.x;
  const float* arow = adj + (long)row * (LL + RR);
  for (int c = t; c < DD; c += 256) {
    float s = 0.f;
    for (int m = 0; m < (LL + RR); ++m) s += arow[m] * graph[((long)b * (LL + RR) + m) * DD + c];
    AG[(long)row * DD + c] = s;
  }
}

// ---------------- host-side orchestration ----------------
extern "C" void kernel_launch(void* const* d_in, const int* in_sizes, int n_in,
                              void* d_out, int out_size, void* d_ws, size_t ws_size,
                              hipStream_t stream)
{
  const float* imgs   = (const float*)d_in[0];
  const float* etexts = (const float*)d_in[1];
  const unsigned char* maskT  = (const unsigned char*)d_in[2];
  const unsigned char* maskTT = (const unsigned char*)d_in[3];
  const unsigned char* maskTV = (const unsigned char*)d_in[4];
  const float* adj    = (const float*)d_in[5];
  const float* answer = (const float*)d_in[8];
  const float* Wt  = (const float*)d_in[9];  const float* bt  = (const float*)d_in[10];
  const float* Wi1 = (const float*)d_in[11]; const float* bi1 = (const float*)d_in[12];
  const float* Wi2 = (const float*)d_in[13]; const float* bi2 = (const float*)d_in[14];
  const float* Wgc = (const float*)d_in[15]; const float* bgc = (const float*)d_in[16];
  const float* Wps = (const float*)d_in[17]; const float* bps = (const float*)d_in[18];
  const float* Wts = (const float*)d_in[19]; const float* bts = (const float*)d_in[20];
  const float* Wsng = (const float*)d_in[21]; const float* bsng = (const float*)d_in[22];
  const float* Wp1 = (const float*)d_in[23]; const float* bp1 = (const float*)d_in[24];
  const float* Wp2 = (const float*)d_in[25]; const float* bp2 = (const float*)d_in[26];
  const float* Wtop = (const float*)d_in[27]; const float* btop = (const float*)d_in[28];
  const float* clues = (const float*)d_in[29];
  const float* Wc1 = (const float*)d_in[30]; const float* bc1 = (const float*)d_in[31];
  const float* Wan1 = (const float*)d_in[32]; const float* ban1 = (const float*)d_in[33];
  const float* Win2 = (const float*)d_in[34]; const float* bin2 = (const float*)d_in[35];
  const float* Wan2 = (const float*)d_in[36]; const float* ban2 = (const float*)d_in[37];
  const float* Wattn = (const float*)d_in[38]; const float* battn = (const float*)d_in[39];
  const float* Wf1 = (const float*)d_in[40]; const float* bf1 = (const float*)d_in[41];
  const float* Wf2 = (const float*)d_in[42]; const float* bf2 = (const float*)d_in[43];
  const float* lng = (const float*)d_in[44]; const float* lnb = (const float*)d_in[45];
  const float* Wgd = (const float*)d_in[46]; const float* bgd = (const float*)d_in[47];
  const float* Wat1 = (const float*)d_in[48]; const float* bat1 = (const float*)d_in[49];
  const float* Wat2 = (const float*)d_in[50]; const float* bat2 = (const float*)d_in[51];
  float* out = (float*)d_out;
  (void)in_sizes; (void)n_in; (void)out_size; (void)ws_size;

  float* ws = (float*)d_ws;
  size_t off = 0;
  auto alloc  = [&](size_t n) { float* p = ws + off; off += n; return p; };
  auto allocB = [&](size_t nbf) { __bf16* p = (__bf16*)(ws + off); off += nbf / 2; return p; };

  float* graph  = alloc((size_t)BB * 144 * DD);
  float* AG     = alloc((size_t)BB * 144 * DD);
  float* imgh   = alloc((size_t)BB * RR * 500);
  float* E_T    = alloc((size_t)BB * LL * DD);
  float* E_V    = alloc((size_t)BB * RR * DD);
  float* scT    = alloc(BB * LL);
  float* scV    = alloc(BB * RR);
  float* scTT   = alloc((size_t)BB * LL * LL);
  float* scTV   = alloc((size_t)BB * LL * RR);
  float* scVV   = alloc((size_t)BB * RR * RR);
  int*   idxbuf = (int*)alloc(512);
  float* Ecat   = alloc((size_t)BB * 50 * DD);
  float* bound  = alloc((size_t)BB * 50 * DD);
  float* lb     = alloc((size_t)BB * 50 * DD);
  float* ans1b  = alloc(ANSN * DD);
  float* ans2b  = alloc(ANSN * DD);
  float* preds  = alloc((size_t)BB * 50 * ANSN * DD);
  float* instl  = alloc((size_t)BB * 50 * DD);
  float* instsc = alloc(BB * 50 * ANSN);
  float* preds2 = alloc((size_t)BB * 50 * ANSN * DD);
  float* qb = alloc((size_t)BB * 50 * DD);
  float* kb = alloc((size_t)BB * 50 * DD);
  float* vb = alloc((size_t)BB * 50 * DD);
  float* ob = alloc((size_t)BB * 50 * DD);
  float* pb = alloc((size_t)BB * 50 * DD);
  float* x1 = alloc((size_t)BB * 50 * DD);
  float* ffh = alloc((size_t)BB * 50 * H1);
  float* ffo = alloc((size_t)BB * 50 * DD);
  float* atoms = alloc(ANSN * BB * 50);
  float* Iemb = alloc(BB * DD);
  float* Temb = alloc(BB * DD);
  float* EIT  = alloc(BB * DD);

  // packed bf16 weights (Kp = ceil32(K), Np = ceil16(N))
  __bf16* pWi1  = allocB((size_t)768 * 512);
  __bf16* pWi2  = allocB((size_t)512 * NP200);
  __bf16* pWt   = allocB((size_t)768 * NP200);
  __bf16* pWsng0 = allocB((size_t)224 * NP200);
  __bf16* pWsng1 = allocB((size_t)224 * NP200);
  __bf16* pWc1  = allocB((size_t)224 * NP200);
  __bf16* pWan1 = allocB((size_t)224 * NP200);
  __bf16* pWan2 = allocB((size_t)224 * NP200);
  __bf16* pWin2 = allocB((size_t)224 * NP200);
  __bf16* pWq   = allocB((size_t)224 * NP200);
  __bf16* pWk   = allocB((size_t)224 * NP200);
  __bf16* pWv   = allocB((size_t)224 * NP200);
  __bf16* pWo   = allocB((size_t)224 * NP200);
  __bf16* pWf1  = allocB((size_t)224 * 400);
  __bf16* pWf2  = allocB((size_t)416 * NP200);
  __bf16* pWgc0 = allocB((size_t)224 * NP200);
  __bf16* pWgc1 = allocB((size_t)224 * NP200);
  __bf16* pP1[3]; __bf16* pP2[3];
  for (int t = 0; t < 3; ++t) { pP1[t] = allocB((size_t)800 * 400); pP2[t] = allocB((size_t)416 * NP200); }

  // ---- one-time weight packing (deterministic; weights constant per call) ----
  k_packW<<<dim3(512, 24), 32, 0, stream>>>(Wi1, 768, 500, 512, pWi1);
  k_packW<<<dim3(NP200, 16), 32, 0, stream>>>(Wi2, 500, DD, NP200, pWi2);
  k_packW<<<dim3(NP200, 24), 32, 0, stream>>>(Wt, 768, DD, NP200, pWt);
  k_packW<<<dim3(NP200, 7), 32, 0, stream>>>(Wsng, DD, DD, NP200, pWsng0);
  k_packW<<<dim3(NP200, 7), 32, 0, stream>>>(Wsng + DD * DD, DD, DD, NP200, pWsng1);
  k_packW<<<dim3(NP200, 7), 32, 0, stream>>>(Wc1, DD, DD, NP200, pWc1);
  k_packW<<<dim3(NP200, 7), 32, 0, stream>>>(Wan1, DD, DD, NP200, pWan1);
  k_packW<<<dim3(NP200, 7), 32, 0, stream>>>(Wan2, DD, DD, NP200, pWan2);
  k_packW<<<dim3(NP200, 7), 32, 0, stream>>>(Win2, DD, DD, NP200, pWin2);
  k_packW<<<dim3(NP200, 7), 32, 0, stream>>>(Wattn, DD, DD, NP200, pWq);
  k_packW<<<dim3(NP200, 7), 32, 0, stream>>>(Wattn + 40000, DD, DD, NP200, pWk);
  k_packW<<<dim3(NP200, 7), 32, 0, stream>>>(Wattn + 80000, DD, DD, NP200, pWv);
  k_packW<<<dim3(NP200, 7), 32, 0, stream>>>(Wattn + 120000, DD, DD, NP200, pWo);
  k_packW<<<dim3(400, 7), 32, 0, stream>>>(Wf1, DD, H1, 400, pWf1);
  k_packW<<<dim3(NP200, 13), 32, 0, stream>>>(Wf2, H1, DD, NP200, pWf2);
  k_packW<<<dim3(NP200, 7), 32, 0, stream>>>(Wgc, DD, DD, NP200, pWgc0);
  k_packW<<<dim3(NP200, 7), 32, 0, stream>>>(Wgc + DD * DD, DD, DD, NP200, pWgc1);
  for (int t = 0; t < 3; ++t) {
    k_packW<<<dim3(400, 25), 32, 0, stream>>>(Wp1 + (size_t)t * 320000, PF, H1, 400, pP1[t]);
    k_packW<<<dim3(NP200, 13), 32, 0, stream>>>(Wp2 + (size_t)t * 80000, H1, DD, NP200, pP2[t]);
  }

  const long GST = 144 * DD;
  float* gtex = graph;
  float* gimg = graph + LL * DD;

  // encoders
  k_gemm<<<dim3(32, 32, 1), 32, 0, stream>>>(imgs, 0, 768, pWi1, 512, bi1, imgh, 0, 500, BB * RR, 768, 500, 1);
  k_gemm<<<dim3(4, 13, BB), 32, 0, stream>>>(imgh, (long)RR * 500, 500, pWi2, NP200, bi2, gimg, GST, DD, RR, 500, DD, 1);
  k_gemm<<<dim3(5, 13, BB), 32, 0, stream>>>(etexts, (long)LL * 768, 768, pWt, NP200, bt, gtex, GST, DD, LL, 768, DD, 0);

  for (int gc = 0; gc < GCN; ++gc) {
    k_gemm<<<dim3(5, 13, BB), 32, 0, stream>>>(gtex, GST, DD, pWsng0, NP200, bsng, E_T, (long)LL * DD, DD, LL, DD, DD, 1);
    k_gemm<<<dim3(4, 13, BB), 32, 0, stream>>>(gimg, GST, DD, pWsng1, NP200, bsng + DD, E_V, (long)RR * DD, DD, RR, DD, DD, 1);
    k_rowdot_relu<<<BB * LL, 64, 0, stream>>>(E_T, Wtop, btop, 0, scT);
    k_rowdot_relu<<<BB * RR, 64, 0, stream>>>(E_V, Wtop + DD, btop, 1, scV);
    k_pair_score<<<dim3(400, 1, BB), 32, 0, stream>>>(gtex, GST, gtex, GST, LL, LL,
        pP1[0], bp1, pP2[0], bp2, Wtop + 2 * DD, btop, 2, scTT);
    k_pair_score<<<dim3(320, 1, BB), 32, 0, stream>>>(gtex, GST, gimg, GST, LL, RR,
        pP1[1], bp1 + 400, pP2[1], bp2 + 200, Wtop + 3 * DD, btop, 3, scTV);
    k_pair_score<<<dim3(256, 1, BB), 32, 0, stream>>>(gimg, GST, gimg, GST, RR, RR,
        pP1[2], bp1 + 800, pP2[2], bp2 + 400, Wtop + 4 * DD, btop, 4, scVV);
    k_topk<<<BB, 256, 0, stream>>>(scT, maskT, LL, idxbuf + 0);
    k_topk<<<BB, 256, 0, stream>>>(scV, nullptr, RR, idxbuf + 80);
    k_topk<<<BB, 256, 0, stream>>>(scTT, maskTT, LL * LL, idxbuf + 160);
    k_topk<<<BB, 256, 0, stream>>>(scTV, maskTV, LL * RR, idxbuf + 240);
    k_topk<<<BB, 256, 0, stream>>>(scVV, nullptr, RR * RR, idxbuf + 320);
    k_gather<<<BB * KK, 64, 0, stream>>>(E_T, LL, idxbuf + 0, Ecat, 0);
    k_gather<<<BB * KK, 64, 0, stream>>>(E_V, RR, idxbuf + 80, Ecat, 10);
    k_pair_rows<<<BB, 32, 0, stream>>>(gtex, GST, gtex, GST, LL, idxbuf + 160,
        pP1[0], bp1, pP2[0], bp2, Ecat, 20);
    k_pair_rows<<<BB, 32, 0, stream>>>(gtex, GST, gimg, GST, RR, idxbuf + 240,
        pP1[1], bp1 + 400, pP2[1], bp2 + 200, Ecat, 30);
    k_pair_rows<<<BB, 32, 0, stream>>>(gimg, GST, gimg, GST, RR, idxbuf + 320,
        pP1[2], bp1 + 800, pP2[2], bp2 + 400, Ecat, 40);
    k_gemm<<<dim3(1, 13, 1), 32, 0, stream>>>(answer, 0, DD, pWan1, NP200, ban1, ans1b, 0, DD, ANSN, DD, DD, 0);
    k_gemm<<<dim3(1, 13, 1), 32, 0, stream>>>(answer, 0, DD, pWan2, NP200, ban2, ans2b, 0, DD, ANSN, DD, DD, 0);
    k_bind<<<BB * 50, 64, 0, stream>>>(Ecat, clues, bound);
    k_gemm<<<dim3(25, 13, 1), 32, 0, stream>>>(bound, 0, DD, pWc1, NP200, bc1, lb, 0, DD, BB * 50, DD, DD, 0);
    k_preds<<<BB * 50, 256, 0, stream>>>(lb, ans1b, preds);
    k_gemm<<<dim3(25, 13, 1), 32, 0, stream>>>(Ecat, 0, DD, pWin2, NP200, bin2, instl, 0, DD, BB * 50, DD, DD, 0);
    k_inst<<<BB * 50, 64, 0, stream>>>(instl, ans2b, instsc);
    for (int a = 0; a < ANSN; ++a) {
      const float* X = preds + a * DD;   // row stride 2*DD
      k_gemm<<<dim3(25, 13, 1), 32, 0, stream>>>(X, 0, 2 * DD, pWq, NP200, battn, qb, 0, DD, BB * 50, DD, DD, 0);
      k_gemm<<<dim3(25, 13, 1), 32, 0, stream>>>(X, 0, 2 * DD, pWk, NP200, battn + DD, kb, 0, DD, BB * 50, DD, DD, 0);
      k_gemm<<<dim3(25, 13, 1), 32, 0, stream>>>(X, 0, 2 * DD, pWv, NP200, battn + 2 * DD, vb, 0, DD, BB * 50, DD, DD, 0);
      k_attn<<<BB * NHEAD, 64, 0, stream>>>(qb, kb, vb, ob);
      k_gemm<<<dim3(25, 13, 1), 32, 0, stream>>>(ob, 0, DD, pWo, NP200, battn + 3 * DD, pb, 0, DD, BB * 50, DD, DD, 0);
      k_ln<<<BB * 50, 64, 0, stream>>>(X, 2 * DD, pb, DD, x1, DD, lng, lnb);
      k_gemm<<<dim3(25, 25, 1), 32, 0, stream>>>(x1, 0, DD, pWf1, 400, bf1, ffh, 0, H1, BB * 50, DD, H1, 1);
      k_gemm<<<dim3(25, 13, 1), 32, 0, stream>>>(ffh, 0, H1, pWf2, NP200, bf2, ffo, 0, DD, BB * 50, H1, DD, 0);
      k_ln<<<BB * 50, 64, 0, stream>>>(x1, DD, ffo, DD, preds2 + a * DD, 2 * DD, lng + DD, lnb + DD);
    }
    k_atoms<<<BB * 50 * ANSN, 256, 0, stream>>>(Ecat, preds, Wat1, bat1, Wat2, bat2, atoms);
    k_softemb<<<BB, 128, 0, stream>>>(graph, LL, RR, Wps, bps, Iemb);
    k_softemb<<<BB, 128, 0, stream>>>(graph, 0, LL, Wts, bts, Temb);
    k_eit<<<BB, 256, 0, stream>>>(Temb, Iemb, Wp1 + 320000, bp1 + 400, Wp2 + 80000, bp2 + 200, EIT);
    k_heads<<<BB, 64, 0, stream>>>(preds2, Wgd, bgd, EIT, instsc, atoms, out + gc * 16);
    k_adjmm<<<BB * 144, 256, 0, stream>>>(adj, graph, AG);
    k_gemm<<<dim3(72, 13, 1), 32, 0, stream>>>(AG, 0, DD, (gc == 0 ? pWgc0 : pWgc1), NP200,
                                               bgc + gc * DD, graph, 0, DD, BB * 144, DD, DD, 1);
  }
}
